// Transformer2_72593537237165
// MI455X (gfx1250) — matrix operand order repaired and checked by reading, not compiled, not run on hardware
//
#include <hip/hip_runtime.h>
#include <math.h>

// ---------------------------------------------------------------------------
// Graph spatio-temporal transformer forward pass for MI455X (gfx1250).
// All dense GEMMs use v_wmma_f32_16x16x32_f16 (fp16 A/B staged in LDS,
// fp32 accumulate), double-buffered LDS ping-pong so global fetch of K-step
// k+1 overlaps the WMMAs of step k.  Small ops (attention T<=9, layernorm,
// glue) are fp32 VALU.
// GEMM requires M%128==0, N%64==0, K%32==0 (true for every call site here).
// Requires ws_size >= ~302 MB (75,497,472 floats).
// ---------------------------------------------------------------------------

typedef __attribute__((ext_vector_type(16))) _Float16 v16h;
typedef __attribute__((ext_vector_type(8)))  _Float16 h8;
typedef __attribute__((ext_vector_type(8)))  float    v8f;
typedef __attribute__((ext_vector_type(4)))  float    f4;

#define GEMM_BM 128
#define GEMM_BN 64
#define GEMM_BK 32
#define LDS_STRIDE 40   // 32 f16 + 8 pad; 80 bytes per row, 16B aligned

// C = act(A @ B + bias).  A: MxK fp32 row-major.
// TRANSB=false: B is KxN row-major.  TRANSB=true: B is NxK row-major (weight W,
// computing X @ W^T).  act: 0=none, 1=relu, 2=sigmoid.  bias may be null.
// 8 waves; wave w computes rows [w*16, w*16+16) x all 64 cols (4 WMMA tiles).
template<bool TRANSB>
__global__ __launch_bounds__(256) void gemm_wmma_kernel(
    const float* __restrict__ A, const float* __restrict__ B,
    const float* __restrict__ bias, float* __restrict__ C,
    int M, int N, int K, int act)
{
    __shared__ __attribute__((aligned(16))) _Float16 As [2][GEMM_BM][LDS_STRIDE];
    __shared__ __attribute__((aligned(16))) _Float16 BsT[2][GEMM_BN][LDS_STRIDE];

    const int tid  = threadIdx.x;
    const int wave = tid >> 5;          // 0..7 -> 16-row strip
    const int lane = tid & 31;
    const int lh = lane >> 4;           // lane half
    const int ll = lane & 15;
    const int rowBase = blockIdx.y * GEMM_BM;
    const int colBase = blockIdx.x * GEMM_BN;

    v8f acc[4] = {{}, {}, {}, {}};

    // staging coordinates (all full tiles; no bounds checks)
    const int ar = tid >> 1;            // 0..127
    const int ac = (tid & 1) << 4;      // 0 or 16
    const int bn_t = tid >> 2;          // 0..63   (TRANSB path)
    const int bk_t = (tid & 3) << 3;    // 0,8,16,24
    const int bk_n = tid >> 3;          // 0..31   (!TRANSB path)
    const int bn_n = (tid & 7) << 3;    // 0..56

    const float* Aptr = A + (size_t)(rowBase + ar) * K + ac;
    const float* Bptr = TRANSB ? B + (size_t)(colBase + bn_t) * K + bk_t
                               : B + (size_t)bk_n * N + (colBase + bn_n);

    // staging registers (global -> regs -> cvt -> LDS)
    f4 ga0, ga1, ga2, ga3, gb0, gb1;

    // ---- prologue: fetch K-step 0 and stage into buffer 0 ----
    {
        const float* ga = Aptr;
        ga0 = *(const f4*)(ga + 0);  ga1 = *(const f4*)(ga + 4);
        ga2 = *(const f4*)(ga + 8);  ga3 = *(const f4*)(ga + 12);
        const float* gb = Bptr;
        gb0 = *(const f4*)(gb + 0);  gb1 = *(const f4*)(gb + 4);
    }
    {
        h8 p0, p1;
#pragma unroll
        for (int j = 0; j < 4; ++j) {
            p0[j] = (_Float16)ga0[j]; p0[j + 4] = (_Float16)ga1[j];
            p1[j] = (_Float16)ga2[j]; p1[j + 4] = (_Float16)ga3[j];
        }
        h8* dst = (h8*)&As[0][ar][ac];
        dst[0] = p0; dst[1] = p1;
        if (TRANSB) {
            h8 p;
#pragma unroll
            for (int j = 0; j < 4; ++j) {
                p[j] = (_Float16)gb0[j]; p[j + 4] = (_Float16)gb1[j];
            }
            *(h8*)&BsT[0][bn_t][bk_t] = p;
        } else {
#pragma unroll
            for (int j = 0; j < 4; ++j) {
                BsT[0][bn_n + j][bk_n]     = (_Float16)gb0[j];
                BsT[0][bn_n + 4 + j][bk_n] = (_Float16)gb1[j];
            }
        }
    }
    __syncthreads();

    int buf = 0;
    for (int k0 = 0; k0 < K; k0 += GEMM_BK) {
        const bool more = (k0 + GEMM_BK) < K;

        // ---- issue global loads for the NEXT K-step (overlaps WMMA) ----
        if (more) {
            const int kn = k0 + GEMM_BK;
            const float* ga = Aptr + kn;
            ga0 = *(const f4*)(ga + 0);  ga1 = *(const f4*)(ga + 4);
            ga2 = *(const f4*)(ga + 8);  ga3 = *(const f4*)(ga + 12);
            const float* gb = TRANSB ? Bptr + kn : Bptr + (size_t)kn * N;
            gb0 = *(const f4*)(gb + 0);  gb1 = *(const f4*)(gb + 4);
        }

        // ---- fragments from LDS[buf] (ISA 7.12.2; all 16B-aligned) ----
        const int m = (wave << 4) + ll;
        h8 a0 = *(const h8*)&As[buf][m][lh << 3];
        h8 a1 = *(const h8*)&As[buf][m][16 + (lh << 3)];
        v16h af = __builtin_shufflevector(a0, a1,
                      0, 1, 2, 3, 4, 5, 6, 7, 8, 9, 10, 11, 12, 13, 14, 15);
        h8 bl0[4], bl1[4];
#pragma unroll
        for (int t = 0; t < 4; ++t) {
            const _Float16* bp = &BsT[buf][(t << 4) + ll][lh << 3];
            bl0[t] = *(const h8*)bp;
            bl1[t] = *(const h8*)(bp + 16);
        }
#pragma unroll
        for (int t = 0; t < 4; ++t) {
            v16h bf = __builtin_shufflevector(bl0[t], bl1[t],
                          0, 1, 2, 3, 4, 5, 6, 7, 8, 9, 10, 11, 12, 13, 14, 15);
            acc[t] = __builtin_amdgcn_wmma_f32_16x16x32_f16(
                false, af, false, bf, (short)0, acc[t], false, false);
        }

        // ---- convert + store next tile into the other buffer ----
        if (more) {
            const int nb = buf ^ 1;
            h8 p0, p1;
#pragma unroll
            for (int j = 0; j < 4; ++j) {
                p0[j] = (_Float16)ga0[j]; p0[j + 4] = (_Float16)ga1[j];
                p1[j] = (_Float16)ga2[j]; p1[j + 4] = (_Float16)ga3[j];
            }
            h8* dst = (h8*)&As[nb][ar][ac];
            dst[0] = p0; dst[1] = p1;
            if (TRANSB) {
                h8 p;
#pragma unroll
                for (int j = 0; j < 4; ++j) {
                    p[j] = (_Float16)gb0[j]; p[j + 4] = (_Float16)gb1[j];
                }
                *(h8*)&BsT[nb][bn_t][bk_t] = p;
            } else {
#pragma unroll
                for (int j = 0; j < 4; ++j) {
                    BsT[nb][bn_n + j][bk_n]     = (_Float16)gb0[j];
                    BsT[nb][bn_n + 4 + j][bk_n] = (_Float16)gb1[j];
                }
            }
        }
        __syncthreads();      // single barrier per K-step (ping-pong)
        buf ^= 1;
    }

    // ---- epilogue: lane element r -> row = r + 8*lh, col tile*16 + ll ----
#pragma unroll
    for (int t = 0; t < 4; ++t) {
        const int col = colBase + (t << 4) + ll;
        float bv = bias ? bias[col] : 0.0f;
#pragma unroll
        for (int r = 0; r < 8; ++r) {
            const int row = rowBase + (wave << 4) + (lh << 3) + r;
            float v = acc[t][r] + bv;
            if (act == 1) v = fmaxf(v, 0.0f);
            else if (act == 2) v = 1.0f / (1.0f + __expf(-v));
            C[(size_t)row * N + col] = v;
        }
    }
}

// ---------------------------------------------------------------------------
// Per-(n,h) attention: q/k/v projected tensors laid out (N, T, H=2, D=64).
// energy = clip(q.k^T * scale, -5, 5); softmax over k; out = att @ v.
// One 64-thread block per (n, h).
// ---------------------------------------------------------------------------
__global__ __launch_bounds__(64) void attention_kernel(
    const float* __restrict__ qp, const float* __restrict__ kp,
    const float* __restrict__ vp, float* __restrict__ out,
    int N, int Tq, int Tk, float scale)
{
    __shared__ float Ks[9][64], Vs[9][64], Qs[9][64], Att[9][9];
    const int n = blockIdx.x >> 1;
    const int h = blockIdx.x & 1;
    const int tid = threadIdx.x;

    for (int i = tid; i < Tk * 64; i += 64) {
        int t = i >> 6, d = i & 63;
        size_t off = (((size_t)n * Tk + t) * 2 + h) * 64 + d;
        Ks[t][d] = kp[off];
        Vs[t][d] = vp[off];
    }
    for (int i = tid; i < Tq * 64; i += 64) {
        int t = i >> 6, d = i & 63;
        Qs[t][d] = qp[(((size_t)n * Tq + t) * 2 + h) * 64 + d];
    }
    __syncthreads();

    for (int i = tid; i < Tq * Tk; i += 64) {
        int q = i / Tk, k = i % Tk;
        float s = 0.0f;
        for (int d = 0; d < 64; ++d) s += Qs[q][d] * Ks[k][d];
        s *= scale;
        Att[q][k] = fminf(fmaxf(s, -5.0f), 5.0f);
    }
    __syncthreads();

    if (tid < Tq) {
        float m = -1e30f;
        for (int k = 0; k < Tk; ++k) m = fmaxf(m, Att[tid][k]);
        float s = 0.0f;
        for (int k = 0; k < Tk; ++k) { float e = __expf(Att[tid][k] - m); Att[tid][k] = e; s += e; }
        float inv = 1.0f / s;
        for (int k = 0; k < Tk; ++k) Att[tid][k] *= inv;
    }
    __syncthreads();

    for (int i = tid; i < Tq * 64; i += 64) {
        int q = i >> 6, d = i & 63;
        float s = 0.0f;
        for (int k = 0; k < Tk; ++k) s += Att[q][k] * Vs[k][d];
        out[(((size_t)n * Tq + q) * 2 + h) * 64 + d] = s;
    }
}

// ---------------------------------------------------------------------------
// LayerNorm over last dim D (64 or 128), optional residual add.
// One wave32 per row, 8 rows per 256-thread block; shfl_xor reductions.
// ---------------------------------------------------------------------------
__global__ __launch_bounds__(256) void layernorm_kernel(
    float* __restrict__ out, const float* __restrict__ x,
    const float* __restrict__ res, const float* __restrict__ g,
    const float* __restrict__ b, int rows, int D)
{
    const int wave = threadIdx.x >> 5, lane = threadIdx.x & 31;
    const int row = blockIdx.x * 8 + wave;
    if (row >= rows) return;
    const float* xr = x + (size_t)row * D;
    const float* rr = res ? res + (size_t)row * D : nullptr;
    const int per = D >> 5;            // 2 or 4
    float vals[4];
    float sum = 0.0f;
    for (int i = 0; i < per; ++i) {
        float v = xr[lane + 32 * i];
        if (rr) v += rr[lane + 32 * i];
        vals[i] = v; sum += v;
    }
    for (int m = 16; m; m >>= 1) sum += __shfl_xor(sum, m, 32);
    float mean = sum / (float)D;
    float vs = 0.0f;
    for (int i = 0; i < per; ++i) { float d = vals[i] - mean; vs += d * d; }
    for (int m = 16; m; m >>= 1) vs += __shfl_xor(vs, m, 32);
    float inv = rsqrtf(vs / (float)D + 1e-5f);
    float* orow = out + (size_t)row * D;
    for (int i = 0; i < per; ++i) {
        int c = lane + 32 * i;
        orow[c] = (vals[i] - mean) * inv * g[c] + b[c];
    }
}

// src1[n][t][c] = (c<64) ? senc[n][t][c] : 0
__global__ void concat_zero_kernel(float* __restrict__ dst,
                                   const float* __restrict__ senc, int total)
{
    int i = blockIdx.x * 256 + threadIdx.x;
    if (i >= total) return;
    int c = i & 127; size_t nt = (size_t)(i >> 7);
    dst[i] = (c < 64) ? senc[nt * 64 + (c & 63)] : 0.0f;
}

// tgt1[n][t][c] = senc[n][6+t][c%64], t in 0..2
__global__ void make_tgt1_kernel(float* __restrict__ dst,
                                 const float* __restrict__ senc, int N)
{
    int i = blockIdx.x * 256 + threadIdx.x;
    int total = N * 3 * 128;
    if (i >= total) return;
    int c = i & 127; int t = (i >> 7) % 3; int n = i / (3 * 128);
    dst[i] = senc[(((size_t)n * 9) + 6 + t) * 64 + (c & 63)];
}

// src2[n][t][c] = (c<64) ? senc[n][t][c] : x1[n][t%3][c-64] - senc[n][t][c-64]
__global__ void make_src2_kernel(float* __restrict__ dst,
                                 const float* __restrict__ senc,
                                 const float* __restrict__ x1, int N)
{
    int i = blockIdx.x * 256 + threadIdx.x;
    int total = N * 9 * 128;
    if (i >= total) return;
    int c = i & 127; int t = (i >> 7) % 9; int n = i / (9 * 128);
    float s = senc[((size_t)n * 9 + t) * 64 + (c & 63)];
    dst[i] = (c < 64) ? s
                      : x1[(((size_t)n * 3) + (t % 3)) * 64 + (c - 64)] - s;
}

// tiny head: out[r][j] = x[r][:] . W[j][:] + b[j], j<2, K=64
__global__ void final_linear_kernel(float* __restrict__ out,
                                    const float* __restrict__ x,
                                    const float* __restrict__ W,
                                    const float* __restrict__ bias, int rows)
{
    int i = blockIdx.x * 256 + threadIdx.x;
    if (i >= rows * 2) return;
    int r = i >> 1, j = i & 1;
    const float* xr = x + (size_t)r * 64;
    float s = bias[j];
    for (int d = 0; d < 64; ++d) s += xr[d] * W[j * 64 + d];
    out[i] = s;
}

// ---------------------------------------------------------------------------
// Host orchestration
// ---------------------------------------------------------------------------
static inline int ceil_div(int a, int b) { return (a + b - 1) / b; }

static void gemm(const float* A, const float* B, const float* bias, float* C,
                 int M, int N, int K, int act, bool transB, hipStream_t s)
{
    dim3 grid(ceil_div(N, GEMM_BN), ceil_div(M, GEMM_BM));
    if (transB) gemm_wmma_kernel<true ><<<grid, 256, 0, s>>>(A, B, bias, C, M, N, K, act);
    else        gemm_wmma_kernel<false><<<grid, 256, 0, s>>>(A, B, bias, C, M, N, K, act);
}

static void layernorm(float* out, const float* x, const float* res,
                      const float* g, const float* b, int rows, int D, hipStream_t s)
{
    layernorm_kernel<<<ceil_div(rows, 8), 256, 0, s>>>(out, x, res, g, b, rows, D);
}

struct TTP {
    const float *wv, *wk, *wq, *wow, *wob, *ln1g, *ln1b, *ln2g, *ln2b,
                *w1w, *w1b, *w2w, *w2b;
};
static TTP make_ttp(void* const* din, int b)
{
    TTP p;
    p.wv   = (const float*)din[b + 0];  p.wk  = (const float*)din[b + 1];
    p.wq   = (const float*)din[b + 2];  p.wow = (const float*)din[b + 3];
    p.wob  = (const float*)din[b + 4];  p.ln1g = (const float*)din[b + 5];
    p.ln1b = (const float*)din[b + 6];  p.ln2g = (const float*)din[b + 7];
    p.ln2b = (const float*)din[b + 8];  p.w1w  = (const float*)din[b + 9];
    p.w1b  = (const float*)din[b + 10]; p.w2w  = (const float*)din[b + 11];
    p.w2b  = (const float*)din[b + 12];
    return p;
}

// _ttransformer: out = LN2( FF( LN1(attn(q,k,v)+q) ) + LN1(...) )
static void run_tt(const TTP& p, const float* value, const float* key,
                   const float* query, int N, int Tq, int Tk,
                   float* outbuf, float* S, hipStream_t s)
{
    const int Rq = N * Tq, Rk = N * Tk;
    float* qp = S;
    float* kp = qp + (size_t)Rq * 128;
    float* vp = kp + (size_t)Rk * 128;
    float* ao = vp + (size_t)Rk * 128;
    float* a  = ao + (size_t)Rq * 128;
    float* x  = a  + (size_t)Rq * 128;
    float* hd = x  + (size_t)Rq * 128;
    float* f  = hd + (size_t)Rq * 512;

    gemm(query, p.wq, nullptr, qp, Rq * 2, 64, 64, 0, true, s);
    gemm(key,   p.wk, nullptr, kp, Rk * 2, 64, 64, 0, true, s);
    gemm(value, p.wv, nullptr, vp, Rk * 2, 64, 64, 0, true, s);
    attention_kernel<<<N * 2, 64, 0, s>>>(qp, kp, vp, ao, N, Tq, Tk,
                                          1.0f / sqrtf(128.0f));
    gemm(ao, p.wow, p.wob, a, Rq, 128, 128, 0, true, s);
    layernorm(x, a, query, p.ln1g, p.ln1b, Rq, 128, s);
    gemm(x,  p.w1w, p.w1b, hd, Rq, 512, 128, 1, true, s);
    gemm(hd, p.w2w, p.w2b, f,  Rq, 128, 512, 0, true, s);
    layernorm(outbuf, f, x, p.ln2g, p.ln2b, Rq, 128, s);
}

static void run_decoder(const TTP& ts, const TTP& tm,
                        const float* ln1g, const float* ln1b,
                        const float* ln2g, const float* ln2b,
                        const float* xenc, const float* tgt1,
                        float* outbuf, float* pool, hipStream_t s)
{
    const int N = 4096, R = N * 3;
    float* t1  = pool + 25165824;        // beyond run_tt scratch high-water
    float* x1b = t1  + (size_t)R * 128;
    float* t2  = x1b + (size_t)R * 128;
    run_tt(ts, tgt1, tgt1, tgt1, N, 3, 3, t1, pool, s);
    layernorm(x1b, t1, tgt1, ln1g, ln1b, R, 128, s);
    run_tt(tm, xenc, xenc, x1b, N, 3, 9, t2, pool, s);
    layernorm(outbuf, t2, x1b, ln2g, ln2b, R, 128, s);
}

extern "C" void kernel_launch(void* const* d_in, const int* in_sizes, int n_in,
                              void* d_out, int out_size, void* d_ws, size_t ws_size,
                              hipStream_t stream)
{
    (void)in_sizes; (void)n_in; (void)out_size; (void)ws_size;
    const int N = 4096, T = 9;
    const int NT   = N * T;          // 36864
    const int NT64 = NT * 64;        // 2,359,296
    const int NT128 = NT * 128;      // 4,718,592

    // -------- input map (setup_inputs() dict insertion order, recursive) ----
    const float* src = (const float*)d_in[0];   // (4096,9,64)
    const float* adj = (const float*)d_in[1];   // (4096,4096)
    const float* l1w = (const float*)d_in[2];  const float* l1b = (const float*)d_in[3];
    const float* l2w = (const float*)d_in[4];  const float* l2b = (const float*)d_in[5];
    const float* l3w = (const float*)d_in[6];  const float* l3b = (const float*)d_in[7];
    TTP ttf  = make_ttp(d_in, 8);               // 8..20
    TTP d1s  = make_ttp(d_in, 21);              // dec1.tt_self 21..33
    TTP d1m  = make_ttp(d_in, 34);              // dec1.tt_mu   34..46
    const float* d1ln1g = (const float*)d_in[47]; const float* d1ln1b = (const float*)d_in[48];
    const float* d1ln2g = (const float*)d_in[49]; const float* d1ln2b = (const float*)d_in[50];
    TTP d2s  = make_ttp(d_in, 51);
    TTP d2m  = make_ttp(d_in, 64);
    const float* d2ln1g = (const float*)d_in[77]; const float* d2ln1b = (const float*)d_in[78];
    const float* d2ln2g = (const float*)d_in[79]; const float* d2ln2b = (const float*)d_in[80];
    const float* n1g = (const float*)d_in[81]; const float* n1b = (const float*)d_in[82];
    const float* n2g = (const float*)d_in[83]; const float* n2b = (const float*)d_in[84];
    const float* fcw = (const float*)d_in[85]; const float* fcb = (const float*)d_in[86];
    const float* lnw = (const float*)d_in[87]; const float* lnb = (const float*)d_in[88];

    // -------- workspace carve (floats) --------------------------------------
    float* W     = (float*)d_ws;
    float* senc  = W;                       // 2,359,296
    float* src1  = senc + NT64;             // 4,718,592 (reused as src2)
    float* tgt1  = src1 + NT128;            // 1,572,864
    float* xenc  = tgt1 + (size_t)N * 3 * 128; // 4,718,592
    float* x1sig = xenc + NT128;            //   786,432 (reused for x2 head)
    float* pool  = x1sig + (size_t)N * 3 * 64; // 61,341,696 scratch pool
    float* ttout  = pool + 51904512;        // transformer output slot
    float* decOut = pool + 29884416;        // decoder output slot

    float* outX  = (float*)d_out;           // (4096,9,128)
    float* outY1 = outX + NT128;            // (4096,3,2)
    float* outY2 = outY1 + N * 3 * 2;

    // -------- Phase A: _graph_sfwt + norm1 ----------------------------------
    float* o1  = pool;
    float* o2  = o1 + NT64;
    float* o3a = o2 + NT64;                 // 36864 x 1024
    float* o3b = o3a + (size_t)NT * 1024;   // 36864 x 512
    gemm(adj, src, nullptr, o1, N, T * 64, N, 1, false, stream);
    gemm(adj, o1,  nullptr, o2, N, T * 64, N, 1, false, stream);
    gemm(o2,  l1w, l1b, o3a, NT, 1024, 64,   1, true, stream);
    gemm(o3a, l2w, l2b, o3b, NT, 512,  1024, 1, true, stream);
    gemm(o3b, l3w, l3b, o1,  NT, 64,   512,  0, true, stream);   // reuse o1
    layernorm(senc, o1, src, n1g, n1b, NT, 64, stream);

    // -------- Phase B: encoder pass 1 ---------------------------------------
    concat_zero_kernel<<<ceil_div(NT128, 256), 256, 0, stream>>>(src1, senc, NT128);
    make_tgt1_kernel<<<ceil_div(N * 3 * 128, 256), 256, 0, stream>>>(tgt1, senc, N);
    run_tt(ttf, src1, src1, src1, N, T, T, ttout, pool, stream);
    layernorm(xenc, ttout, src1, n2g, n2b, NT, 128, stream);

    // -------- Phase C: decoder 1 + sigmoid head + y1 ------------------------
    run_decoder(d1s, d1m, d1ln1g, d1ln1b, d1ln2g, d1ln2b,
                xenc, tgt1, decOut, pool, stream);
    gemm(decOut, fcw, fcb, x1sig, N * 3, 64, 128, 2, true, stream); // sigmoid
    final_linear_kernel<<<ceil_div(N * 3 * 2, 256), 256, 0, stream>>>(
        outY1, x1sig, lnw, lnb, N * 3);

    // -------- Phase D: src2, encoder pass 2 (-> d_out x), decoder 2, y2 -----
    make_src2_kernel<<<ceil_div(NT128, 256), 256, 0, stream>>>(src1, senc, x1sig, N);
    run_tt(ttf, src1, src1, src1, N, T, T, ttout, pool, stream);
    layernorm(outX, ttout, src1, n2g, n2b, NT, 128, stream);       // output x
    run_decoder(d2s, d2m, d2ln1g, d2ln1b, d2ln2g, d2ln2b,
                outX, tgt1, decOut, pool, stream);
    gemm(decOut, fcw, fcb, x1sig, N * 3, 64, 128, 2, true, stream);
    final_linear_kernel<<<ceil_div(N * 3 * 2, 256), 256, 0, stream>>>(
        outY2, x1sig, lnw, lnb, N * 3);
}
